// Attention_59158879535719
// MI455X (gfx1250) — matrix operand order repaired and checked by reading, not compiled, not run on hardware
//
#include <hip/hip_runtime.h>
#include <hip/hip_bf16.h>

// ---------------------------------------------------------------------------
// MI455X (gfx1250) fused attention block:
//   K0: pre-convert x / w_qkv (q rows pre-scaled) / w_out to bf16
//   K1: qkv = Wqkv * x   (bf16 WMMA; q,v stored transposed)
//   K2: flash attention per (b,head), bf16 WMMA, online softmax,
//       K/V tiles staged to LDS via async-to-LDS (triple buffer) when the
//       builtins exist, else reg-staged double buffer
//   K3: out = Wout * attn_out + bias (f32 out)
// ---------------------------------------------------------------------------

typedef __attribute__((ext_vector_type(16))) __bf16 v16bf;
typedef __attribute__((ext_vector_type(8)))  __bf16 v8bf;
typedef __attribute__((ext_vector_type(4)))  __bf16 v4bf;
typedef __attribute__((ext_vector_type(2)))  __bf16 v2bf;
typedef __attribute__((ext_vector_type(8)))  float  v8f;
typedef __attribute__((ext_vector_type(4)))  float  v4f;
typedef __attribute__((ext_vector_type(4)))  int    v4i;

#define SEQ   4096
#define CIN   256
#define HID   512
#define NHEAD 8
#define DHEAD 64

#define KLD   40   // padded LDS row stride for K tile (16B aligned, low-conflict)
#define VLD   72   // padded LDS row stride for V tile (16B aligned, low-conflict)

#if __has_builtin(__builtin_amdgcn_global_load_async_to_lds_b128) && \
    __has_builtin(__builtin_amdgcn_s_wait_asynccnt)
#define USE_ASYNC 1
#else
#define USE_ASYNC 0
#endif

// ---- f32 -> bf16 conversion ------------------------------------------------
#if __has_builtin(__builtin_amdgcn_cvt_pk_bf16_f32)
__device__ __forceinline__ v2bf cvt2(float a, float b) {
  return __builtin_amdgcn_cvt_pk_bf16_f32(a, b);
}
#else
__device__ __forceinline__ v2bf cvt2(float a, float b) {
  const unsigned ua = __builtin_bit_cast(unsigned, a) + 0x8000u;
  const unsigned ub = __builtin_bit_cast(unsigned, b) + 0x8000u;
  // dst = { ub[31:16], ua[31:16] }
  return __builtin_bit_cast(v2bf, __builtin_amdgcn_perm(ub, ua, 0x07060302u));
}
#endif

__device__ __forceinline__ v8bf cvt8(v4f a, v4f b) {
  v2bf p0 = cvt2(a[0], a[1]), p1 = cvt2(a[2], a[3]);
  v2bf p2 = cvt2(b[0], b[1]), p3 = cvt2(b[2], b[3]);
  v4bf q0 = __builtin_shufflevector(p0, p1, 0, 1, 2, 3);
  v4bf q1 = __builtin_shufflevector(p2, p3, 0, 1, 2, 3);
  return __builtin_shufflevector(q0, q1, 0, 1, 2, 3, 4, 5, 6, 7);
}

__device__ __forceinline__ v8bf cvt8v(v8f s) {
  return cvt8(__builtin_shufflevector(s, s, 0, 1, 2, 3),
              __builtin_shufflevector(s, s, 4, 5, 6, 7));
}

__device__ __forceinline__ v16bf cat8(v8bf lo, v8bf hi) {
  return __builtin_shufflevector(lo, hi, 0,1,2,3,4,5,6,7,8,9,10,11,12,13,14,15);
}

// ---- WMMA fragment loaders (layouts per ISA 7.12.2) -----------------------
__device__ __forceinline__ v16bf load_a_bf16(const unsigned short* base, int ld,
                                             int row0, int k0, int lane) {
  const unsigned short* p =
      base + (size_t)(row0 + (lane & 15)) * ld + k0 + ((lane >> 4) << 3);
  return cat8(*(const v8bf*)p, *(const v8bf*)(p + 16));
}

__device__ __forceinline__ v16bf load_b_bf16(const unsigned short* base, int ld,
                                             int k0, int n0, int lane) {
  const unsigned short* p =
      base + (size_t)(k0 + ((lane >> 4) << 3)) * ld + n0 + (lane & 15);
  v16bf r;
#pragma unroll
  for (int s = 0; s < 16; s++)
    r[s] = __builtin_bit_cast(__bf16, p[(size_t)((s & 7) + ((s >> 3) << 4)) * ld]);
  return r;
}

__device__ __forceinline__ v8f wmma_bf16(v16bf a, v16bf b, v8f c) {
  return __builtin_amdgcn_wmma_f32_16x16x32_bf16(false, a, false, b,
                                                 (short)0, c, false, false);
}

#if USE_ASYNC
// per-lane 16B global -> LDS async copy (tracked by ASYNCcnt).
// Builtin signature (from hipcc diagnostic): param 1 is int4 addrspace(1)*,
// so cast pointers precisely (through void* to shed const).
__device__ __forceinline__ void async_ld16(const unsigned short* g,
                                           unsigned short* l) {
  typedef v4i __attribute__((address_space(1)))* gp_t;
  typedef v4i __attribute__((address_space(3)))* lp_t;
  __builtin_amdgcn_global_load_async_to_lds_b128((gp_t)(void*)g,
                                                 (lp_t)(void*)l, 0, 0);
}
#endif

// ---------------------------------------------------------------------------
// K0: elementwise f32 -> bf16 (8 elems/thread); first `nscaled` elems * 0.125
// ---------------------------------------------------------------------------
__global__ __launch_bounds__(256)
void cvt_kernel(const float* __restrict__ src, unsigned short* __restrict__ dst,
                int n8, int nscaled) {
  const int i = blockIdx.x * 256 + threadIdx.x;
  if (i >= n8) return;
  const v4f* p = (const v4f*)(src + (size_t)i * 8);
  v4f a = p[0], b = p[1];
  if (i * 8 < nscaled) { a *= 0.125f; b *= 0.125f; }
  *(v8bf*)(dst + (size_t)i * 8) = cvt8(a, b);
}

// ---------------------------------------------------------------------------
// K1: qkv projection. grid = (S/128, 1536/128, b). 8 waves: 2(M) x 4(N),
// wave tile 64x32, K = 256. q -> qT[b,h,s,d] (scale baked into weights),
// k -> kN[b,hd,s], v -> vT[b,h,s,d].
// ---------------------------------------------------------------------------
__global__ __launch_bounds__(256)
void qkv_kernel(const unsigned short* __restrict__ xbf,
                const unsigned short* __restrict__ wbf,
                unsigned short* __restrict__ qT, unsigned short* __restrict__ kN,
                unsigned short* __restrict__ vT) {
  const int b = blockIdx.z;
  const int lane = threadIdx.x & 31, wave = threadIdx.x >> 5;
  const int m0 = blockIdx.y * 128 + (wave >> 2) * 64;
  const int n0 = blockIdx.x * 128 + (wave & 3) * 32;
  const unsigned short* xb = xbf + (size_t)b * CIN * SEQ;
  const int region = m0 / HID;   // 0=q, 1=k, 2=v (uniform per block tile)

  v8f acc[4][2] = {};
  for (int k0 = 0; k0 < CIN; k0 += 32) {
    v16bf af[4], bf[2];
#pragma unroll
    for (int i = 0; i < 4; i++) af[i] = load_a_bf16(wbf, CIN, m0 + 16 * i, k0, lane);
#pragma unroll
    for (int j = 0; j < 2; j++) bf[j] = load_b_bf16(xb, SEQ, k0, n0 + 16 * j, lane);
#pragma unroll
    for (int i = 0; i < 4; i++)
#pragma unroll
      for (int j = 0; j < 2; j++) acc[i][j] = wmma_bf16(af[i], bf[j], acc[i][j]);
  }

  const int col = lane & 15, hh = lane >> 4;
#pragma unroll
  for (int i = 0; i < 4; i++)
#pragma unroll
    for (int j = 0; j < 2; j++) {
      const int n = n0 + 16 * j + col;
      const int obase = m0 + 16 * i + 8 * hh;          // rows obase..obase+7
      if (region == 0) {                               // q: transposed
        const int head = obase >> 6, d0 = obase & 63;
        *(v8bf*)(qT + (((size_t)b * NHEAD + head) * SEQ + n) * DHEAD + d0) =
            cvt8v(acc[i][j]);
      } else if (region == 1) {                        // k: natural [hd, s]
        const size_t base = ((size_t)b * HID + (obase - HID)) * SEQ + n;
#pragma unroll
        for (int r = 0; r < 8; r += 2) {
          const unsigned u = __builtin_bit_cast(unsigned,
              cvt2(acc[i][j][r], acc[i][j][r + 1]));
          kN[base + (size_t)r * SEQ]       = (unsigned short)u;
          kN[base + (size_t)(r + 1) * SEQ] = (unsigned short)(u >> 16);
        }
      } else {                                         // v: transposed
        const int oo = obase - 2 * HID;
        const int head = oo >> 6, d0 = oo & 63;
        *(v8bf*)(vT + (((size_t)b * NHEAD + head) * SEQ + n) * DHEAD + d0) =
            cvt8v(acc[i][j]);
      }
    }
}

// ---------------------------------------------------------------------------
// K2: flash attention. grid = (S/128, b*NHEAD). Each wave owns 16 query rows.
// K/V tiles for each 32-key block are staged once per WG into LDS.
// Async path: triple-buffered, issued one block ahead, s_wait_asynccnt(2)
// before the barrier (async loads complete in order).
// Fallback: reg-staged double buffer.
// ---------------------------------------------------------------------------
__global__ __launch_bounds__(256)
void attn_kernel(const unsigned short* __restrict__ qT,
                 const unsigned short* __restrict__ kN,
                 const unsigned short* __restrict__ vT,
                 unsigned short* __restrict__ aoT) {
#if USE_ASYNC
  __shared__ __align__(16) unsigned short kbuf[3][64 * KLD];
  __shared__ __align__(16) unsigned short vbuf[3][32 * VLD];
#else
  __shared__ __align__(16) unsigned short kbuf[2][64 * KLD];
  __shared__ __align__(16) unsigned short vbuf[2][32 * VLD];
#endif
  __shared__ __align__(16) unsigned short pbuf[8][16 * 32];   // 1 KB per wave

  const int bh = blockIdx.y;                 // b*NHEAD + head
  const int tid = threadIdx.x;
  const int lane = tid & 31, wave = tid >> 5;
  const int i0 = blockIdx.x * 128 + wave * 16;
  const int b = bh >> 3, head = bh & 7;

  const unsigned short* q  = qT + (size_t)bh * SEQ * DHEAD;
  const unsigned short* kk = kN + (size_t)bh * DHEAD * SEQ;
  const unsigned short* vv = vT + (size_t)bh * SEQ * DHEAD;

  v16bf qa[2];
  qa[0] = load_a_bf16(q, DHEAD, i0, 0, lane);
  qa[1] = load_a_bf16(q, DHEAD, i0, 32, lane);

  v16bf ones;
#pragma unroll
  for (int i = 0; i < 16; i++)
    ones[i] = __builtin_bit_cast(__bf16, (unsigned short)0x3F80);  // 1.0bf

  v8f acc[4] = {};
  v8f lacc = {};
  float mrow[8];
#pragma unroll
  for (int r = 0; r < 8; r++) mrow[r] = -1e30f;

  const float LOG2E = 1.4426950408889634f;
  unsigned short* pl = &pbuf[wave][0];
  const int col = lane & 15, hh = lane >> 4;
  // permuted key row for V's B-fragment: physical K = lane -> key offset
  const int keyperm = (lane >> 1) + ((lane & 1) << 4);

  // staging assignments: K tile 64x32 (4 x 16B per row), V tile 32x64 (8 x 16B)
  const int krow = tid >> 2, kq = tid & 3;     // K: row 0..63, quarter
  const int vrow = tid >> 3, vq = tid & 7;     // V: row 0..31, eighth
  const int koff = krow * KLD + (kq << 3);
  const int voff = vrow * VLD + (vq << 3);
  const unsigned short* ksrc = kk + (size_t)krow * SEQ + (kq << 3);

#if USE_ASYNC
  async_ld16(ksrc, &kbuf[0][koff]);                                  // block 0
  async_ld16(vv + (size_t)vrow * DHEAD + (vq << 3), &vbuf[0][voff]);
  int cur = 0;
#else
  uint4 kreg = *(const uint4*)ksrc;                                    // j0 = 0
  uint4 vreg = *(const uint4*)(vv + (size_t)vrow * DHEAD + (vq << 3)); // j0 = 0
#endif

  for (int jb = 0; jb < SEQ / 32; jb++) {
#if USE_ASYNC
    const int nxt = (cur == 2) ? 0 : cur + 1;
    const int jn = (jb + 1) & (SEQ / 32 - 1);  // wrap keeps issue count uniform
    async_ld16(ksrc + jn * 32, &kbuf[nxt][koff]);
    async_ld16(vv + (size_t)(jn * 32 + vrow) * DHEAD + (vq << 3), &vbuf[nxt][voff]);
    __builtin_amdgcn_s_wait_asynccnt(2);       // current block's copies done
    __syncthreads();
    unsigned short* kb = &kbuf[cur][0];
    unsigned short* vb = &vbuf[cur][0];
    cur = nxt;
#else
    const int nb = jb & 1;
    unsigned short* kb = &kbuf[nb][0];
    unsigned short* vb = &vbuf[nb][0];
    *(uint4*)(kb + koff) = kreg;
    *(uint4*)(vb + voff) = vreg;
    __syncthreads();
    if (jb + 1 < SEQ / 32) {                   // prefetch next block into regs
      kreg = *(const uint4*)(ksrc + (jb + 1) * 32);
      vreg = *(const uint4*)(vv + (size_t)((jb + 1) * 32 + vrow) * DHEAD + (vq << 3));
    }
#endif

    // sim tiles: S0 (cols 0..15 of block), S1 (cols 16..31), K over d = 64
    v8f s0 = {}, s1 = {};
    s0 = wmma_bf16(qa[0], load_b_bf16(kb, KLD, 0,  0,  lane), s0);
    s0 = wmma_bf16(qa[1], load_b_bf16(kb, KLD, 32, 0,  lane), s0);
    s1 = wmma_bf16(qa[0], load_b_bf16(kb, KLD, 0,  16, lane), s1);
    s1 = wmma_bf16(qa[1], load_b_bf16(kb, KLD, 32, 16, lane), s1);

    // online softmax; row r lives in lanes 0-15 (VGPR r) / 16-31 (row r+8)
    float afac[8];
#pragma unroll
    for (int r = 0; r < 8; r++) {
      float mx = fmaxf(s0[r], s1[r]);
#pragma unroll
      for (int off = 8; off >= 1; off >>= 1)
        mx = fmaxf(mx, __shfl_xor(mx, off, 32));
      const float mn = fmaxf(mrow[r], mx);
      afac[r] = exp2f((mrow[r] - mn) * LOG2E);
      mrow[r] = mn;
      const float p0 = exp2f((s0[r] - mn) * LOG2E);
      const float p1 = exp2f((s1[r] - mn) * LOG2E);
      // C-layout -> LDS P tile, K-permuted: key col -> K=2*col,
      // key 16+col -> K=2*col+1  => one aligned b32 store per row
      const int row = r + 8 * hh;
      *(unsigned*)(pl + row * 32 + 2 * col) =
          __builtin_bit_cast(unsigned, cvt2(p0, p1));
    }

    // rescale accumulators only when some lane's running max moved
    float amin = fminf(fminf(fminf(afac[0], afac[1]), fminf(afac[2], afac[3])),
                       fminf(fminf(afac[4], afac[5]), fminf(afac[6], afac[7])));
    if (amin < 1.0f) {
#pragma unroll
      for (int r = 0; r < 8; r++) {
#pragma unroll
        for (int c = 0; c < 4; c++) acc[c][r] *= afac[r];
        lacc[r] *= afac[r];
      }
    }

    __builtin_amdgcn_wave_barrier();   // keep DS stores before DS loads
    const v16bf pa = load_a_bf16(pl, 32, 0, 0, lane);

    // out[i, d] += P * V^T (permuted rows); denom: lacc += P * 1
#pragma unroll
    for (int c = 0; c < 4; c++) {
      const unsigned short* p = vb + c * 16 + col;
      v16bf vf;
#pragma unroll
      for (int s = 0; s < 16; s++) {
        const int kp = 8 * hh + (s & 7) + ((s >> 3) << 4);
        vf[s] = __builtin_bit_cast(__bf16, p[((kp >> 1) + ((kp & 1) << 4)) * VLD]);
      }
      acc[c] = wmma_bf16(pa, vf, acc[c]);
    }
    lacc = wmma_bf16(pa, ones, lacc);
  }

  // normalize and store transposed: aoT[b, hd, s]; rows r are consecutive s
  float inv[8];
#pragma unroll
  for (int r = 0; r < 8; r++) inv[r] = 1.0f / lacc[r];
#pragma unroll
  for (int c = 0; c < 4; c++) {
    v8f s;
#pragma unroll
    for (int r = 0; r < 8; r++) s[r] = acc[c][r] * inv[r];
    const size_t hd = (size_t)b * HID + head * DHEAD + c * 16 + col;
    *(v8bf*)(aoT + hd * SEQ + i0 + 8 * hh) = cvt8v(s);
  }
}

// ---------------------------------------------------------------------------
// K3: output projection + bias. grid = (S/128, 256/128, b). K = 512.
// ---------------------------------------------------------------------------
__global__ __launch_bounds__(256)
void proj_kernel(const unsigned short* __restrict__ aoT,
                 const unsigned short* __restrict__ wbf,
                 const float* __restrict__ bias, float* __restrict__ out) {
  const int b = blockIdx.z;
  const int lane = threadIdx.x & 31, wave = threadIdx.x >> 5;
  const int m0 = blockIdx.y * 128 + (wave >> 2) * 64;
  const int n0 = blockIdx.x * 128 + (wave & 3) * 32;
  const unsigned short* aob = aoT + (size_t)b * HID * SEQ;

  v8f acc[4][2] = {};
  for (int k0 = 0; k0 < HID; k0 += 32) {
    v16bf af[4], bf[2];
#pragma unroll
    for (int i = 0; i < 4; i++) af[i] = load_a_bf16(wbf, HID, m0 + 16 * i, k0, lane);
#pragma unroll
    for (int j = 0; j < 2; j++) bf[j] = load_b_bf16(aob, SEQ, k0, n0 + 16 * j, lane);
#pragma unroll
    for (int i = 0; i < 4; i++)
#pragma unroll
      for (int j = 0; j < 2; j++) acc[i][j] = wmma_bf16(af[i], bf[j], acc[i][j]);
  }

  const int col = lane & 15, hh = lane >> 4;
#pragma unroll
  for (int i = 0; i < 4; i++)
#pragma unroll
    for (int j = 0; j < 2; j++) {
      const int n = n0 + 16 * j + col;
#pragma unroll
      for (int r = 0; r < 8; r++) {
        const int o = m0 + 16 * i + r + 8 * hh;
        out[((size_t)b * CIN + o) * SEQ + n] = acc[i][j][r] + bias[o];
      }
    }
}

// ---------------------------------------------------------------------------
extern "C" void kernel_launch(void* const* d_in, const int* in_sizes, int n_in,
                              void* d_out, int out_size, void* d_ws, size_t ws_size,
                              hipStream_t stream) {
  (void)in_sizes; (void)n_in; (void)out_size; (void)ws_size;
  const float* x     = (const float*)d_in[0];
  const float* w_qkv = (const float*)d_in[1];
  const float* w_out = (const float*)d_in[2];
  const float* b_out = (const float*)d_in[3];

  // workspace (bf16 ushort elems):
  //   qT | kN | vT | aoT : each 2*512*4096 = 4,194,304
  //   xbf: 2*256*4096 = 2,097,152 ; wqkvbf: 1536*256 = 393,216 ; woutbf: 131,072
  unsigned short* qT     = (unsigned short*)d_ws;
  unsigned short* kN     = qT  + (size_t)2 * HID * SEQ;
  unsigned short* vT     = kN  + (size_t)2 * HID * SEQ;
  unsigned short* aoT    = vT  + (size_t)2 * HID * SEQ;
  unsigned short* xbf    = aoT + (size_t)2 * HID * SEQ;
  unsigned short* wqkvbf = xbf + (size_t)2 * CIN * SEQ;
  unsigned short* woutbf = wqkvbf + (size_t)3 * HID * CIN;
  float* out = (float*)d_out;

  const int nx = 2 * CIN * SEQ / 8, nw = 3 * HID * CIN / 8, no = CIN * HID / 8;
  cvt_kernel<<<dim3((nx + 255) / 256), 256, 0, stream>>>(x, xbf, nx, 0);
  cvt_kernel<<<dim3((nw + 255) / 256), 256, 0, stream>>>(w_qkv, wqkvbf, nw,
                                                         HID * CIN);  // q rows *0.125
  cvt_kernel<<<dim3((no + 255) / 256), 256, 0, stream>>>(w_out, woutbf, no, 0);

  qkv_kernel <<<dim3(SEQ / 128, 1536 / 128, 2), 256, 0, stream>>>(xbf, wqkvbf, qT, kN, vT);
  attn_kernel<<<dim3(SEQ / 128, 2 * NHEAD),      256, 0, stream>>>(qT, kN, vT, aoT);
  proj_kernel<<<dim3(SEQ / 128, CIN / 128, 2),   256, 0, stream>>>(aoT, woutbf, b_out, out);
}
